// SwinLikeBlock_53360673686141
// MI455X (gfx1250) — matrix operand order repaired and checked by reading, not compiled, not run on hardware
//
#include <hip/hip_runtime.h>
#include <hip/hip_bf16.h>
#include <math.h>

// ---------------------------------------------------------------------------
// Swin-like block, fused for MI455X (gfx1250, wave32, WMMA bf16 16x16x32).
//   x: (16,224,224,128) fp32 ; 16384 windows of 49 tokens (padded to 64 rows).
//   Pass A: per-window [LN1 + shift-gather -> QKV -> attn -> proj -> +shortcut]
//   Pass B: per-64-token [LN2 -> fc1+GELU -> fc2 -> +residual], in-place.
// HBM-bound (~2.5 GB traffic); all GEMMs on v_wmma_f32_16x16x32_bf16.
// All GEMMs are computed in transposed-output form (A = N-major weights,
// B = activation^T): operand fragments stay two contiguous 16-byte loads per lane AND the
// 8 D-elements per lane become 8 consecutive channels of one token, so every
// store is a 128-bit vector op (ds_store_b128 / global b128 RMW).
// ---------------------------------------------------------------------------

typedef __bf16 bf16;
typedef __attribute__((ext_vector_type(16))) __bf16 v16bf;
typedef __attribute__((ext_vector_type(8)))  __bf16 v8bf;
typedef __attribute__((ext_vector_type(8)))  float  v8f;

#define SHIFT 3
#define HDIM  224
#define CH    128
#define NWIN  16384   // 16 * 32 * 32
#define NTOK  49
#define MPAD  64
#define HID   256

__device__ __forceinline__ float wave_sum(float v) {
#pragma unroll
  for (int m = 16; m >= 1; m >>= 1) v += __shfl_xor(v, m, 32);
  return v;
}
__device__ __forceinline__ float wave_max(float v) {
#pragma unroll
  for (int m = 16; m >= 1; m >>= 1) v = fmaxf(v, __shfl_xor(v, m, 32));
  return v;
}

// A fragment (16x32 MxK), row-major source, ld elements, 16B-aligned rows.
// lane<16: K = {0..7, 16..23}; lane>=16: K = {8..15, 24..31}. (ISA 7.12.2)
__device__ __forceinline__ v16bf load_a_frag(const bf16* p, int ld, int lane) {
  const int row = lane & 15, hf = lane >> 4;
  const bf16* r0 = p + row * ld + hf * 8;
  v8bf lo = *(const v8bf*)(r0);
  v8bf hi = *(const v8bf*)(r0 + 16);
  return __builtin_shufflevector(lo, hi, 0, 1, 2, 3, 4, 5, 6, 7,
                                 8, 9, 10, 11, 12, 13, 14, 15);
}
// B fragment (32x16 KxN) taken as the transpose of an N x K row-major source:
// per lane two 16-byte loads.  lane = N column; lane<16: K = {0..7, 16..23},
// lane>=16: K = {8..15, 24..31}.
__device__ __forceinline__ v16bf load_bT_frag(const bf16* p, int ld, int lane) {
  const int col = lane & 15, hf = lane >> 4;
  const bf16* r0 = p + col * ld + hf * 8;
  v8bf lo = *(const v8bf*)(r0);
  v8bf hi = *(const v8bf*)(r0 + 16);
  return __builtin_shufflevector(lo, hi, 0, 1, 2, 3, 4, 5, 6, 7,
                                 8, 9, 10, 11, 12, 13, 14, 15);
}
__device__ __forceinline__ v8f wmma_bf16(v16bf a, v16bf b, v8f c) {
  return __builtin_amdgcn_wmma_f32_16x16x32_bf16(false, a, false, b,
                                                 (short)0, c, false, false);
}

// ------------------ weight conversion (fp32 -> bf16, transposed) -----------
// ws layout (bf16):
//   [0      , 49152) wqT : [384][128]   (qkv_w^T)
//   [49152  , 65536) wpT : [128][128]   (proj_w^T)
//   [65536  , 98304) w1T : [256][128]   (fc1_w^T)
//   [98304  ,131072) w2T : [128][256]   (fc2_w^T)
__global__ void cvt_weights_kernel(const float* __restrict__ qkv_w,
                                   const float* __restrict__ proj_w,
                                   const float* __restrict__ fc1_w,
                                   const float* __restrict__ fc2_w,
                                   bf16* __restrict__ out) {
  int i = blockIdx.x * blockDim.x + threadIdx.x;
  if (i < 49152) {
    int n = i >> 7, k = i & 127;
    out[i] = (bf16)qkv_w[k * 384 + n];
  } else if (i < 65536) {
    int j = i - 49152, n = j >> 7, k = j & 127;
    out[i] = (bf16)proj_w[k * 128 + n];
  } else if (i < 98304) {
    int j = i - 65536, n = j >> 7, k = j & 127;
    out[i] = (bf16)fc1_w[k * 256 + n];
  } else if (i < 131072) {
    int j = i - 98304, n = j >> 8, k = j & 255;
    out[i] = (bf16)fc2_w[k * 128 + n];
  }
}

// --------------------------- fused window attention ------------------------
// One 256-thread block (8 waves) per window. Dynamic LDS (80 KB):
//   [0      ,16384) : 64x128 bf16 x-tile (reused: 64x64 f32 scores/bf16 probs)
//   [16384  ,49152) : 64x256 bf16 Q|K
//   [49152  ,65536) : 128x64 bf16 V^T  (channel-major for contiguous A-frags)
//   [65536  ,81920) : 64x128 bf16 attn-out
__global__ __launch_bounds__(256)
void swin_attn_kernel(const float* __restrict__ x,
                      const float* __restrict__ g1, const float* __restrict__ b1,
                      const bf16*  __restrict__ wqT, const float* __restrict__ bq,
                      const bf16*  __restrict__ wpT, const float* __restrict__ bp,
                      float* __restrict__ xres) {
  extern __shared__ char smem[];
  bf16*  s_x   = (bf16*)smem;
  bf16*  s_qk  = (bf16*)(smem + 16384);
  bf16*  s_vT  = (bf16*)(smem + 49152);
  bf16*  s_out = (bf16*)(smem + 65536);
  float* s_sc  = (float*)smem;           // overlay (x-tile dead after QKV)
  bf16*  s_pr  = (bf16*)smem;            // probs overlay (scores live in regs)

  const int widx = blockIdx.x;
  const int b = widx >> 10, rem = widx & 1023, wh = rem >> 5, ww = rem & 31;
  const int tid = threadIdx.x, wave = tid >> 5, lane = tid & 31;
  const int col16 = lane & 15, hf = lane >> 4;

  // ---- stage 0: gather shifted window + LN1, fp32 -> bf16 LDS tile --------
  for (int tok = wave; tok < MPAD; tok += 8) {
    const int c0 = lane * 4;
    bf16* dst = s_x + tok * CH + c0;
    if (tok < NTOK) {
      const int hi = wh * 7 + tok / 7, wi = ww * 7 + tok % 7;
      const int si = (hi + SHIFT) % HDIM, sj = (wi + SHIFT) % HDIM;
      const float* src = x + (((size_t)b * HDIM + si) * HDIM + sj) * CH;
      float4 v = *(const float4*)(src + c0);
      float mean = wave_sum(v.x + v.y + v.z + v.w) * (1.0f / CH);
      float dx = v.x - mean, dy = v.y - mean, dz = v.z - mean, dw = v.w - mean;
      float var = wave_sum(dx * dx + dy * dy + dz * dz + dw * dw) * (1.0f / CH);
      float r = rsqrtf(var + 1e-5f);
      dst[0] = (bf16)(dx * r * g1[c0 + 0] + b1[c0 + 0]);
      dst[1] = (bf16)(dy * r * g1[c0 + 1] + b1[c0 + 1]);
      dst[2] = (bf16)(dz * r * g1[c0 + 2] + b1[c0 + 2]);
      dst[3] = (bf16)(dw * r * g1[c0 + 3] + b1[c0 + 3]);
    } else {
      dst[0] = dst[1] = dst[2] = dst[3] = (bf16)0.0f;   // pad rows 49..63
    }
  }
  __syncthreads();

  // ---- stage 1a: (Q|K)^T tiles = wq^T @ xn^T -> s_qk row-major [64][256] --
  // 64 tiles; tb fixed per wave (B-frags loop-invariant); D elements are 8
  // consecutive channels of one token -> single v8bf store.
  for (int t = wave; t < 64; t += 8) {
    const int nb = t >> 2, tb = t & 3;
    v8f acc = {};
#pragma unroll
    for (int ks = 0; ks < 4; ++ks) {
      v16bf a  = load_a_frag(wqT + (nb * 16) * CH + ks * 32, CH, lane);
      v16bf bb = load_bT_frag(s_x + tb * 16 * CH + ks * 32, CH, lane);
      acc = wmma_bf16(a, bb, acc);
    }
    const int tok = tb * 16 + col16;
    const float* bias8 = bq + nb * 16 + hf * 8;
    v8bf pk;
#pragma unroll
    for (int r8 = 0; r8 < 8; ++r8) pk[r8] = (bf16)(acc[r8] + bias8[r8]);
    *(v8bf*)(s_qk + tok * 256 + nb * 16 + hf * 8) = pk;
  }
  // ---- stage 1b: V = xn @ wq[:,256:384] + b -> s_vT [128][64] (transposed)-
  // normal orientation; transposed store is contiguous over r8 -> b128 store.
  for (int t = wave; t < 32; t += 8) {
    const int c0 = (t >> 2) * 16, mt = t & 3;   // c0: v-channel block
    v8f acc = {};
#pragma unroll
    for (int ks = 0; ks < 4; ++ks) {
      v16bf a  = load_a_frag(s_x + mt * 16 * CH + ks * 32, CH, lane);
      v16bf bb = load_bT_frag(wqT + (256 + c0) * CH + ks * 32, CH, lane);
      acc = wmma_bf16(a, bb, acc);
    }
    const float bias = bq[256 + c0 + col16];
    v8bf pk;
#pragma unroll
    for (int r8 = 0; r8 < 8; ++r8) pk[r8] = (bf16)(acc[r8] + bias);
    *(v8bf*)(s_vT + (c0 + col16) * MPAD + mt * 16 + hf * 8) = pk;
  }
  __syncthreads();

  // ---- stage 2: per-head  softmax(QK^T / sqrt(32)) @ V --------------------
  const float scale = 0.17677669529663687f;  // 1/sqrt(32)
  for (int h = 0; h < 4; ++h) {
    // scores^T tiles: A = K rows, B = Q^T; D[k][q] stored into row-major
    // s_sc[q][k] -> 8 consecutive f32 per lane -> one v8f store.
    for (int t = wave; t < 16; t += 8) {
      const int kb = t >> 2, qb = t & 3;
      v16bf a  = load_a_frag(s_qk + kb * 16 * 256 + CH + h * 32, 256, lane);
      v16bf bb = load_bT_frag(s_qk + qb * 16 * 256 + h * 32, 256, lane);
      v8f acc = {};
      acc = wmma_bf16(a, bb, acc);
      const int qtok = qb * 16 + col16;
      v8f sc = acc * scale;
      *(v8f*)(s_sc + qtok * 64 + kb * 16 + hf * 8) = sc;
    }
    __syncthreads();
    // softmax: wave owns 8 rows, lane covers cols {lane, lane+32}
    float p0[8], p1[8];
#pragma unroll
    for (int rr = 0; rr < 8; ++rr) {
      const int row = wave * 8 + rr;
      const bool v1 = (lane + 32) < NTOK;
      float s0 = s_sc[row * 64 + lane];
      float s1 = v1 ? s_sc[row * 64 + lane + 32] : -3.0e38f;
      float mx = wave_max(fmaxf(s0, s1));
      float e0 = expf(s0 - mx);
      float e1 = v1 ? expf(s1 - mx) : 0.0f;
      float inv = 1.0f / wave_sum(e0 + e1);
      p0[rr] = e0 * inv; p1[rr] = e1 * inv;
    }
    __syncthreads();
#pragma unroll
    for (int rr = 0; rr < 8; ++rr) {
      const int row = wave * 8 + rr;
      s_pr[row * 64 + lane]      = (bf16)p0[rr];
      s_pr[row * 64 + lane + 32] = (bf16)p1[rr];
    }
    __syncthreads();
    // out^T tiles: A = V_h^T rows (s_vT), B = P^T (s_pr row-major);
    // D[ch][q] -> 8 consecutive channels of one token -> v8bf store.
    {
      const int cb = wave >> 2, qb = wave & 3;   // 2 x 4 = 8 tiles
      v8f acc = {};
#pragma unroll
      for (int ks = 0; ks < 2; ++ks) {
        v16bf a  = load_a_frag(s_vT + (h * 32 + cb * 16) * MPAD + ks * 32,
                               MPAD, lane);
        v16bf bb = load_bT_frag(s_pr + qb * 16 * 64 + ks * 32, 64, lane);
        acc = wmma_bf16(a, bb, acc);
      }
      const int tok = qb * 16 + col16;
      v8bf pk;
#pragma unroll
      for (int r8 = 0; r8 < 8; ++r8) pk[r8] = (bf16)acc[r8];
      *(v8bf*)(s_out + tok * CH + h * 32 + cb * 16 + hf * 8) = pk;
    }
    __syncthreads();
  }

  // ---- stage 3: proj^T + bias + shortcut, inverse-roll scatter ------------
  // A = wp^T rows, B = attn-out^T; per lane: one token, 8 consecutive
  // channels -> two float4 global loads + two float4 stores (128-bit RMW).
  for (int t = wave; t < 32; t += 8) {
    const int nb = t >> 2, tb = t & 3;
    v8f acc = {};
#pragma unroll
    for (int ks = 0; ks < 4; ++ks) {
      v16bf a  = load_a_frag(wpT + (nb * 16) * CH + ks * 32, CH, lane);
      v16bf bb = load_bT_frag(s_out + tb * 16 * CH + ks * 32, CH, lane);
      acc = wmma_bf16(a, bb, acc);
    }
    const int tok = tb * 16 + col16;
    if (tok < NTOK) {                       // uniform per lane
      const int hi = wh * 7 + tok / 7, wi = ww * 7 + tok % 7;
      const int oi = (hi + SHIFT) % HDIM, oj = (wi + SHIFT) % HDIM;
      const int c0 = nb * 16 + hf * 8;
      const float* bias8 = bp + c0;
      float* dst = xres + (((size_t)b * HDIM + oi) * HDIM + oj) * CH + c0;
      const float* sc = x + (((size_t)b * HDIM + oi) * HDIM + oj) * CH + c0;
      float4 r0 = *(const float4*)(sc);
      float4 r1 = *(const float4*)(sc + 4);
      float4 o0 = make_float4(r0.x + acc[0] + bias8[0],
                              r0.y + acc[1] + bias8[1],
                              r0.z + acc[2] + bias8[2],
                              r0.w + acc[3] + bias8[3]);
      float4 o1 = make_float4(r1.x + acc[4] + bias8[4],
                              r1.y + acc[5] + bias8[5],
                              r1.z + acc[6] + bias8[6],
                              r1.w + acc[7] + bias8[7]);
      *(float4*)(dst)     = o0;
      *(float4*)(dst + 4) = o1;
    }
  }
}

// --------------------------- fused MLP (in-place) --------------------------
__global__ __launch_bounds__(256)
void swin_mlp_kernel(float* __restrict__ xio,
                     const float* __restrict__ g2, const float* __restrict__ b2,
                     const bf16*  __restrict__ w1T, const float* __restrict__ f1b,
                     const bf16*  __restrict__ w2T, const float* __restrict__ f2b) {
  __shared__ bf16 s_xn[MPAD * CH];    // 16 KB
  __shared__ bf16 s_h[MPAD * HID];    // 32 KB
  const int tid = threadIdx.x, wave = tid >> 5, lane = tid & 31;
  const int col16 = lane & 15, hf = lane >> 4;
  const size_t tokbase = (size_t)blockIdx.x * MPAD;

  // LN2 -> bf16 tile
  for (int tok = wave; tok < MPAD; tok += 8) {
    const int c0 = lane * 4;
    const float* src = xio + (tokbase + tok) * CH;
    float4 v = *(const float4*)(src + c0);
    float mean = wave_sum(v.x + v.y + v.z + v.w) * (1.0f / CH);
    float dx = v.x - mean, dy = v.y - mean, dz = v.z - mean, dw = v.w - mean;
    float var = wave_sum(dx * dx + dy * dy + dz * dz + dw * dw) * (1.0f / CH);
    float r = rsqrtf(var + 1e-5f);
    bf16* dst = s_xn + tok * CH + c0;
    dst[0] = (bf16)(dx * r * g2[c0 + 0] + b2[c0 + 0]);
    dst[1] = (bf16)(dy * r * g2[c0 + 1] + b2[c0 + 1]);
    dst[2] = (bf16)(dz * r * g2[c0 + 2] + b2[c0 + 2]);
    dst[3] = (bf16)(dw * r * g2[c0 + 3] + b2[c0 + 3]);
  }
  __syncthreads();

  // fc1^T (A = w1^T rows, B = xn^T) + exact GELU -> s_h row-major [64][256]
  for (int t = wave; t < 64; t += 8) {
    const int hb = t >> 2, tb = t & 3;
    v8f acc = {};
#pragma unroll
    for (int ks = 0; ks < 4; ++ks) {
      v16bf a  = load_a_frag(w1T + (hb * 16) * CH + ks * 32, CH, lane);
      v16bf bb = load_bT_frag(s_xn + tb * 16 * CH + ks * 32, CH, lane);
      acc = wmma_bf16(a, bb, acc);
    }
    const int tok = tb * 16 + col16;
    const float* bias8 = f1b + hb * 16 + hf * 8;
    v8bf pk;
#pragma unroll
    for (int r8 = 0; r8 < 8; ++r8) {
      const float v = acc[r8] + bias8[r8];
      pk[r8] = (bf16)(0.5f * v * (1.0f + erff(v * 0.70710678118654752f)));
    }
    *(v8bf*)(s_h + tok * HID + hb * 16 + hf * 8) = pk;
  }
  __syncthreads();

  // fc2^T (A = w2^T rows, B = h^T) + bias + residual: 128-bit in-place RMW
  for (int t = wave; t < 32; t += 8) {
    const int nb = t >> 2, tb = t & 3;
    v8f acc = {};
#pragma unroll
    for (int ks = 0; ks < 8; ++ks) {
      v16bf a  = load_a_frag(w2T + (nb * 16) * HID + ks * 32, HID, lane);
      v16bf bb = load_bT_frag(s_h + tb * 16 * HID + ks * 32, HID, lane);
      acc = wmma_bf16(a, bb, acc);
    }
    const int tok = tb * 16 + col16;
    const int c0 = nb * 16 + hf * 8;
    const float* bias8 = f2b + c0;
    float* dst = xio + (tokbase + tok) * CH + c0;
    float4 r0 = *(const float4*)(dst);
    float4 r1 = *(const float4*)(dst + 4);
    float4 o0 = make_float4(r0.x + acc[0] + bias8[0],
                            r0.y + acc[1] + bias8[1],
                            r0.z + acc[2] + bias8[2],
                            r0.w + acc[3] + bias8[3]);
    float4 o1 = make_float4(r1.x + acc[4] + bias8[4],
                            r1.y + acc[5] + bias8[5],
                            r1.z + acc[6] + bias8[6],
                            r1.w + acc[7] + bias8[7]);
    *(float4*)(dst)     = o0;
    *(float4*)(dst + 4) = o1;
  }
}

// ---------------------------------------------------------------------------
extern "C" void kernel_launch(void* const* d_in, const int* in_sizes, int n_in,
                              void* d_out, int out_size, void* d_ws, size_t ws_size,
                              hipStream_t stream) {
  (void)in_sizes; (void)n_in; (void)out_size; (void)ws_size;
  const float* x      = (const float*)d_in[0];
  const float* g1     = (const float*)d_in[1];
  const float* b1     = (const float*)d_in[2];
  const float* qkv_w  = (const float*)d_in[3];
  const float* qkv_b  = (const float*)d_in[4];
  const float* proj_w = (const float*)d_in[5];
  const float* proj_b = (const float*)d_in[6];
  const float* g2     = (const float*)d_in[7];
  const float* b2     = (const float*)d_in[8];
  const float* fc1_w  = (const float*)d_in[9];
  const float* fc1_b  = (const float*)d_in[10];
  const float* fc2_w  = (const float*)d_in[11];
  const float* fc2_b  = (const float*)d_in[12];
  float* out = (float*)d_out;

  bf16* wbuf = (bf16*)d_ws;           // 131072 bf16 = 256 KB scratch
  bf16* wqT = wbuf;                   // [384][128]
  bf16* wpT = wbuf + 49152;           // [128][128]
  bf16* w1T = wbuf + 65536;           // [256][128]
  bf16* w2T = wbuf + 98304;           // [128][256]

  cvt_weights_kernel<<<512, 256, 0, stream>>>(qkv_w, proj_w, fc1_w, fc2_w, wbuf);

  // 80 KB dynamic LDS per block (up to 4 blocks per 320 KB WGP)
  hipFuncSetAttribute((const void*)swin_attn_kernel,
                      hipFuncAttributeMaxDynamicSharedMemorySize, 81920);
  swin_attn_kernel<<<NWIN, 256, 81920, stream>>>(x, g1, b1, wqT, qkv_b, wpT,
                                                 proj_b, out);

  swin_mlp_kernel<<<12544, 256, 0, stream>>>(out, g2, b2, w1T, fc1_b, w2T,
                                             fc2_b);
}
